// CausalSelfAttention_352187319178
// MI455X (gfx1250) — matrix operand order repaired and checked by reading, not compiled, not run on hardware
//
#include <hip/hip_runtime.h>

typedef __bf16 bf16;
typedef __attribute__((ext_vector_type(16))) __bf16 v16bf;
typedef __attribute__((ext_vector_type(8)))  __bf16 v8bf;
typedef __attribute__((ext_vector_type(8)))  float  v8f;

#define WMMA_BF16(a, b, c) \
  __builtin_amdgcn_wmma_f32_16x16x32_bf16(false, (a), false, (b), (short)0, (c), false, false)

__device__ __forceinline__ v16bf pack16(v8bf lo, v8bf hi) {
  v16bf r;
#pragma unroll
  for (int i = 0; i < 8; ++i) { r[i] = lo[i]; r[i + 8] = hi[i]; }
  return r;
}

// A 16x32 fragment (lane = row M, K-chunks at half*8 and 16+half*8) from K-contiguous row
__device__ __forceinline__ v16bf a_frag(const bf16* rowp, int kk, int half) {
  return pack16(*(const v8bf*)(rowp + kk + half * 8),
                *(const v8bf*)(rowp + kk + 16 + half * 8));
}
// B 32x16 fragment (lane = col N, K-chunks at half*8 and 16+half*8) from K-contiguous row
__device__ __forceinline__ v16bf b_frag(const bf16* rowp, int kk, int half) {
  return pack16(*(const v8bf*)(rowp + kk + half * 8),
                *(const v8bf*)(rowp + kk + 16 + half * 8));
}

// CDNA5 async global->LDS copy (ASYNCcnt-tracked, no VGPR staging).
// vdst = LDS byte offset (low 32 bits of generic LDS pointer), vaddr = 64-bit global addr.
__device__ __forceinline__ void async_ld16(unsigned lds, const bf16* g) {
  asm volatile("global_load_async_to_lds_b128 %0, %1, off" :: "v"(lds), "v"(g) : "memory");
}
__device__ __forceinline__ void wait_async0() {
#if __has_builtin(__builtin_amdgcn_s_wait_asynccnt)
  __builtin_amdgcn_s_wait_asynccnt(0);
#else
  asm volatile("s_wait_asynccnt 0x0" ::: "memory");
#endif
}

// ---- fp32 -> bf16 conversion (optionally scaling the first scale_n elements) ----
__global__ void __launch_bounds__(256) cvt_kernel(const float* __restrict__ src,
                                                  bf16* __restrict__ dst, int n,
                                                  int scale_n, float s) {
  int i = blockIdx.x * blockDim.x + threadIdx.x;
  if (i < n) {
    float v = src[i];
    if (i < scale_n) v *= s;
    dst[i] = (bf16)v;
  }
}

// ---- LDS-tiled GEMM: C[M,N] = A[M,1024] @ B[N,1024]^T, block tile 128x128, BK=64 ----
// Staging via async global->LDS (double-buffered, one barrier per K-step).
// MODE 0: QKV epilogue (scatter to q/k head-major, V transposed). MODE 1: f32 store.
template <int MODE>
__global__ void __launch_bounds__(256) gemm128(
    const bf16* __restrict__ Am, const bf16* __restrict__ Bm,
    bf16* __restrict__ qb, bf16* __restrict__ kb, bf16* __restrict__ vtb,
    float* __restrict__ outf, int ntiles) {
  __shared__ __align__(128) bf16 As[2][128 * 72];  // 72-pitch: conflict-free frag reads
  __shared__ __align__(128) bf16 Bs[2][128 * 72];

  const int t = threadIdx.x;
  const int lane = t & 31, half = lane >> 4, lrow = lane & 15, w = t >> 5;
  const int mt = blockIdx.x / ntiles, ntb = blockIdx.x % ntiles;
  const int m0 = mt * 128, n0 = ntb * 128;

  // staging: thread t owns a contiguous 64B segment of one row (coalesced)
  const int srow = t >> 1, scol = (t & 1) * 32;
  const bf16* agp = Am + (size_t)(m0 + srow) * 1024 + scol;
  const bf16* bgp = Bm + (size_t)(n0 + srow) * 1024 + scol;
  const int sbase = srow * 72 + scol;
  const unsigned asl[2] = {(unsigned)(size_t)&As[0][sbase], (unsigned)(size_t)&As[1][sbase]};
  const unsigned bsl[2] = {(unsigned)(size_t)&Bs[0][sbase], (unsigned)(size_t)&Bs[1][sbase]};

#pragma unroll
  for (int c = 0; c < 4; ++c) {  // prologue: tile 0 into buffer 0
    async_ld16(asl[0] + c * 16, agp + c * 8);
    async_ld16(bsl[0] + c * 16, bgp + c * 8);
  }
  wait_async0();
  __syncthreads();

  v8f acc[8] = {};
#pragma unroll 2
  for (int it = 0; it < 16; ++it) {
    const int buf = it & 1;
    if (it < 15) {  // async-fetch next K-tile into the other buffer (overlaps WMMAs)
      const int k0 = (it + 1) * 64;
#pragma unroll
      for (int c = 0; c < 4; ++c) {
        async_ld16(asl[buf ^ 1] + c * 16, agp + k0 + c * 8);
        async_ld16(bsl[buf ^ 1] + c * 16, bgp + k0 + c * 8);
      }
    }
    const bf16* arow = &As[buf][(w * 16 + lrow) * 72];
#pragma unroll
    for (int kk = 0; kk < 64; kk += 32) {
      v16bf a = a_frag(arow, kk, half);
#pragma unroll
      for (int nt = 0; nt < 8; ++nt) {
        v16bf b = b_frag(&Bs[buf][(nt * 16 + lrow) * 72], kk, half);
        acc[nt] = WMMA_BF16(a, b, acc[nt]);
      }
    }
    wait_async0();
    __syncthreads();
  }

  if (MODE == 1) {
    // f32 direct store: lanes lrow -> 64B contiguous runs
#pragma unroll
    for (int nt = 0; nt < 8; ++nt) {
      int n = n0 + nt * 16 + lrow;
#pragma unroll
      for (int r = 0; r < 8; ++r) {
        int m = m0 + w * 16 + r + 8 * half;
        outf[(size_t)m * 1024 + n] = acc[nt][r];
      }
    }
  } else {
    const int third = n0 >> 10;  // tile is 128-wide, thirds are 1024-aligned
    if (third < 2) {
      bf16* dstb = (third == 0) ? qb : kb;
      const int nb = n0 - third * 1024;
#pragma unroll
      for (int nt = 0; nt < 8; ++nt) {
        int n = nb + nt * 16 + lrow;
        int hh = n >> 6, d = n & 63;
#pragma unroll
        for (int r = 0; r < 8; ++r) {
          int m = m0 + w * 16 + r + 8 * half;
          int bb = m >> 11, s = m & 2047;
          dstb[(((bb * 16 + hh) * 2048) + s) * 64 + d] = (bf16)acc[nt][r];
        }
      }
    } else {
      // V -> vT[b,h,d,s]: lane holds fixed d, its 8 regs are 8 consecutive s.
      // Packing regs into v8bf IS the transpose: one 16B contiguous store per lane.
#pragma unroll
      for (int nt = 0; nt < 8; ++nt) {
        int f = n0 - 2048 + nt * 16 + lrow;  // feature = h*64 + d
        int hh = f >> 6, d = f & 63;
        int m = m0 + w * 16 + 8 * half;      // first of 8 consecutive s
        int bb = m >> 11, s = m & 2047;
        v8bf pk;
#pragma unroll
        for (int r = 0; r < 8; ++r) pk[r] = (bf16)acc[nt][r];
        *(v8bf*)(vtb + (size_t)((bb * 16 + hh) * 64 + d) * 2048 + s) = pk;
      }
    }
  }
}

// ---- Causal flash attention: block = 4 waves = 64 query rows of one (b,h) ----
// K/V tiles staged cooperatively into LDS (coalesced), shared by all 4 waves.
// Score scale (1/sqrt(64)) is pre-folded into the Q weights at conversion time.
__global__ void __launch_bounds__(128) attn_kernel(
    const bf16* __restrict__ q, const bf16* __restrict__ k,
    const bf16* __restrict__ vt, bf16* __restrict__ attnb) {
  __shared__ __align__(128) bf16 Ks[32 * 72];       // [key][d], pitch 72
  __shared__ __align__(128) bf16 Vs[64 * 40];       // [d][key], pitch 40
  __shared__ __align__(128) bf16 Ps[4 * 16 * 40];   // per-wave P tile, pitch 40

  const int t = threadIdx.x;
  const int lane = t & 31, half = lane >> 4, lrow = lane & 15, w = t >> 5;
  const int qblk = (blockIdx.x & 31) * 64;
  const int h = (blockIdx.x >> 5) & 15;
  const int b = blockIdx.x >> 9;
  const int q0 = qblk + w * 16;

  const bf16* qh = q + (size_t)((b * 16 + h) * 2048) * 64;
  const bf16* kh = k + (size_t)((b * 16 + h) * 2048) * 64;
  const bf16* vh = vt + (size_t)((b * 16 + h) * 64) * 2048;
  bf16* pl = Ps + w * 16 * 40;

  // staging addresses: contiguous 32B per thread (coalesced)
  const bf16* kgp = kh + (size_t)(t >> 2) * 64 + (t & 3) * 16;    // + j0*64
  const bf16* vgp = vh + (size_t)(t >> 1) * 2048 + (t & 1) * 16;  // + j0
  bf16* ksp = &Ks[(t >> 2) * 72 + (t & 3) * 16];
  bf16* vsp = &Vs[(t >> 1) * 40 + (t & 1) * 16];

  v16bf qa0 = a_frag(qh + (q0 + lrow) * 64, 0, half);
  v16bf qa1 = a_frag(qh + (q0 + lrow) * 64, 32, half);

  float Mrow[8], Lrow[8];
  v8f O0 = {}, O1 = {}, O2 = {}, O3 = {};
#pragma unroll
  for (int r = 0; r < 8; ++r) { Mrow[r] = -3.0e38f; Lrow[r] = 0.0f; }

  const int jmax = qblk + 63;

  for (int j0 = 0; j0 <= jmax; j0 += 32) {
    // issue next tile's global loads before the barrier (overlap prior compute)
    v8bf kr0 = *(const v8bf*)(kgp + (size_t)j0 * 64);
    v8bf kr1 = *(const v8bf*)(kgp + (size_t)j0 * 64 + 8);
    v8bf vr0 = *(const v8bf*)(vgp + j0);
    v8bf vr1 = *(const v8bf*)(vgp + j0 + 8);
    __syncthreads();
    *(v8bf*)ksp = kr0; *(v8bf*)(ksp + 8) = kr1;
    *(v8bf*)vsp = vr0; *(v8bf*)(vsp + 8) = vr1;
    __syncthreads();

    if (j0 <= q0 + 15) {
      v8f z = {};
      v8f sc0 = WMMA_BF16(qa0, b_frag(&Ks[lrow * 72], 0, half), z);
      sc0     = WMMA_BF16(qa1, b_frag(&Ks[lrow * 72], 32, half), sc0);
      v8f sc1 = WMMA_BF16(qa0, b_frag(&Ks[(16 + lrow) * 72], 0, half), z);
      sc1     = WMMA_BF16(qa1, b_frag(&Ks[(16 + lrow) * 72], 32, half), sc1);

      float p0[8], p1[8], rmax[8], rsum[8], alpha[8];
      const int key0 = j0 + lrow, key1 = key0 + 16;
#pragma unroll
      for (int r = 0; r < 8; ++r) {
        int qrow = q0 + r + 8 * half;
        float s0 = sc0[r]; if (key0 > qrow) s0 = -3.0e38f;
        float s1 = sc1[r]; if (key1 > qrow) s1 = -3.0e38f;
        p0[r] = s0; p1[r] = s1;
        rmax[r] = fmaxf(s0, s1);
      }
#pragma unroll
      for (int off = 1; off < 16; off <<= 1)
#pragma unroll
        for (int r = 0; r < 8; ++r)
          rmax[r] = fmaxf(rmax[r], __shfl_xor(rmax[r], off, 32));

#pragma unroll
      for (int r = 0; r < 8; ++r) {
        float mnew = fmaxf(Mrow[r], rmax[r]);
        alpha[r] = __expf(Mrow[r] - mnew);
        Mrow[r] = mnew;
        float e0 = __expf(p0[r] - mnew);
        float e1 = __expf(p1[r] - mnew);
        rsum[r] = e0 + e1;
        pl[(r + 8 * half) * 40 + lrow] = (bf16)e0;
        pl[(r + 8 * half) * 40 + 16 + lrow] = (bf16)e1;
      }
#pragma unroll
      for (int off = 1; off < 16; off <<= 1)
#pragma unroll
        for (int r = 0; r < 8; ++r)
          rsum[r] += __shfl_xor(rsum[r], off, 32);
#pragma unroll
      for (int r = 0; r < 8; ++r) Lrow[r] = Lrow[r] * alpha[r] + rsum[r];

      // P back in A-fragment layout (DS ops in-order within a wave)
      v16bf pa = pack16(*(const v8bf*)(pl + lrow * 40 + half * 8),
                        *(const v8bf*)(pl + lrow * 40 + 16 + half * 8));

#pragma unroll
      for (int r = 0; r < 8; ++r) {
        O0[r] *= alpha[r]; O1[r] *= alpha[r]; O2[r] *= alpha[r]; O3[r] *= alpha[r];
      }
      O0 = WMMA_BF16(pa, b_frag(&Vs[(0  + lrow) * 40], 0, half), O0);
      O1 = WMMA_BF16(pa, b_frag(&Vs[(16 + lrow) * 40], 0, half), O1);
      O2 = WMMA_BF16(pa, b_frag(&Vs[(32 + lrow) * 40], 0, half), O2);
      O3 = WMMA_BF16(pa, b_frag(&Vs[(48 + lrow) * 40], 0, half), O3);
    }
  }

#pragma unroll
  for (int r = 0; r < 8; ++r) {
    float inv = 1.0f / Lrow[r];
    int row = q0 + r + 8 * half;
    bf16* dst = attnb + (size_t)(b * 2048 + row) * 1024 + h * 64 + lrow;
    dst[0]  = (bf16)(O0[r] * inv);
    dst[16] = (bf16)(O1[r] * inv);
    dst[32] = (bf16)(O2[r] * inv);
    dst[48] = (bf16)(O3[r] * inv);
  }
}

extern "C" void kernel_launch(void* const* d_in, const int* in_sizes, int n_in,
                              void* d_out, int out_size, void* d_ws, size_t ws_size,
                              hipStream_t stream) {
  const float* x     = (const float*)d_in[0];  // [4,2048,1024]
  const float* w_qkv = (const float*)d_in[1];  // [3072,1024]
  const float* w_out = (const float*)d_in[2];  // [1024,1024]
  float* out = (float*)d_out;                  // [4,2048,1024] f32

  char* ws = (char*)d_ws;
  bf16* xb    = (bf16*)(ws);                   // 16 MB
  bf16* wqkvb = (bf16*)(ws + 16777216);        //  6 MB
  bf16* woutb = (bf16*)(ws + 23068672);        //  2 MB
  bf16* qb    = (bf16*)(ws + 25165824);        // 16 MB [b,h,s,d]
  bf16* kb    = (bf16*)(ws + 41943040);        // 16 MB [b,h,s,d]
  bf16* vtb   = (bf16*)(ws + 58720256);        // 16 MB [b,h,d,s]
  bf16* attnb = (bf16*)(ws + 75497472);        // 16 MB [b,s,h*d]  (total 88 MB)

  // x: plain convert; w_qkv: fold 1/sqrt(hd)=0.125 into the Q rows (first 1024*1024 elems)
  cvt_kernel<<<(8388608 + 255) / 256, 256, 0, stream>>>(x, xb, 8388608, 0, 1.0f);
  cvt_kernel<<<(3145728 + 255) / 256, 256, 0, stream>>>(w_qkv, wqkvb, 3145728, 1048576, 0.125f);
  cvt_kernel<<<(1048576 + 255) / 256, 256, 0, stream>>>(w_out, woutb, 1048576, 0, 1.0f);

  // QKV: (8192/128) x (3072/128) = 64 x 24 blocks
  gemm128<0><<<64 * 24, 256, 0, stream>>>(xb, wqkvb, qb, kb, vtb, nullptr, 24);
  // attention: 4*16*32 = 2048 blocks of 4 waves (64 query rows each)
  attn_kernel<<<2048, 128, 0, stream>>>(qb, kb, vtb, attnb);
  // out-proj: 64 x 8 blocks
  gemm128<1><<<64 * 8, 256, 0, stream>>>(attnb, woutb, nullptr, nullptr, nullptr, out, 8);
}
